// MyGNN_30829275250962
// MI455X (gfx1250) — matrix operand order repaired and checked by reading, not compiled, not run on hardware
//
#include <hip/hip_runtime.h>

typedef __attribute__((ext_vector_type(16))) _Float16 v16h;
typedef __attribute__((ext_vector_type(8)))  float    v8f;

#define BATCH 64
#define NPART 512
#define NFEAT 30
#define NOUT  64
#define NDENSE 128
#define M_TOT (BATCH * NPART)   // 32768
#define XCAT_STRIDE 160         // padded K stride for xcat (covers 129 -> 160, 67 -> 96)

// ---------------------------------------------------------------------------
// Workspace offsets (bytes)
// ---------------------------------------------------------------------------
static constexpr size_t OFF_MASK = 0;                               // B*N f32      = 131072
static constexpr size_t OFF_S    = OFF_MASK + 131072;               // B*N f32
static constexpr size_t OFF_T    = OFF_S    + 131072;               // B*N f32
static constexpr size_t OFF_X    = OFF_T    + 131072;               // B*N*64 f32   = 8388608
static constexpr size_t OFF_XCAT = OFF_X    + 8388608;              // B*N*160 f16  = 10485760
static constexpr size_t OFF_W16  = OFF_XCAT + 10485760;             // 64*160 f16   = 20480 (transposed)
static constexpr size_t OFF_XFIN = OFF_W16  + 20480;                // B*N*64 f16   = 4194304
static constexpr size_t OFF_DW0  = OFF_XFIN + 4194304;              // 128*32768 f16= 8388608 (transposed)
static constexpr size_t OFF_H0   = OFF_DW0  + 8388608;              // 64*128 f16
static constexpr size_t OFF_DW1  = OFF_H0   + 16384;                // 128*128 f16 (transposed)
static constexpr size_t OFF_H1   = OFF_DW1  + 32768;                // 64*128 f16

// ---------------------------------------------------------------------------
// K0: embeddings + concat -> x (B*N x 64 f32, features 33..63 zeroed), mask
// ---------------------------------------------------------------------------
__global__ void prep_kernel(const float* __restrict__ xx,
                            const float* __restrict__ emb1,
                            const float* __restrict__ emb2,
                            const float* __restrict__ emb3,
                            float* __restrict__ x, float* __restrict__ mask) {
    int m = blockIdx.x * blockDim.x + threadIdx.x;
    if (m >= M_TOT) return;
    const float* row = xx + (size_t)m * NFEAT;
    float mk = row[0];
    mask[m] = mk;
    int i1 = (int)fabsf(row[27]);
    int i2 = (int)fabsf(row[28]);
    int i3 = (int)fabsf(row[29]);
    float* o = x + (size_t)m * 64;
    o[0] = emb1[i1 * 2 + 0];  o[1] = emb1[i1 * 2 + 1];
    o[2] = emb2[i2 * 2 + 0];  o[3] = emb2[i2 * 2 + 1];
    o[4] = emb3[i3 * 2 + 0];  o[5] = emb3[i3 * 2 + 1];
    #pragma unroll
    for (int f = 0; f < 27; ++f) o[6 + f] = row[f];
    #pragma unroll
    for (int f = 33; f < 64; ++f) o[f] = 0.0f;
}

// ---------------------------------------------------------------------------
// K1: per-batch pairwise RBF sums. One block per batch, coords staged in LDS.
// S[i] = sum_j exp(-10*d2(i,j)),  T[i] = sum_j d2*w   (unmasked, per reference)
// ---------------------------------------------------------------------------
__global__ __launch_bounds__(256) void pairwise_kernel(const float* __restrict__ x,
                                                       int coordOff, int ndim,
                                                       float* __restrict__ S,
                                                       float* __restrict__ T) {
    __shared__ float cs[NPART * 4];
    int b = blockIdx.x;
    const float* xb = x + (size_t)b * NPART * 64;
    for (int j = threadIdx.x; j < NPART; j += 256) {
        #pragma unroll
        for (int d = 0; d < 4; ++d)
            cs[j * 4 + d] = (d < ndim) ? xb[(size_t)j * 64 + coordOff + d] : 0.0f;
    }
    __syncthreads();
    for (int i = threadIdx.x; i < NPART; i += 256) {
        float c0 = cs[i*4+0], c1 = cs[i*4+1], c2 = cs[i*4+2], c3 = cs[i*4+3];
        float s = 0.0f, t = 0.0f;
        for (int j = 0; j < NPART; ++j) {
            float d0 = cs[j*4+0] - c0;
            float d1 = cs[j*4+1] - c1;
            float d2 = cs[j*4+2] - c2;
            float d3 = cs[j*4+3] - c3;
            float dist = d0*d0 + d1*d1 + d2*d2 + d3*d3;
            float w = __expf(-10.0f * dist);
            s += w;
            t += dist * w;
        }
        S[b * NPART + i] = s;
        T[b * NPART + i] = t;
    }
}

// ---------------------------------------------------------------------------
// K2: build xcat = [x, x*(mask*S)-x, mask*T, 0-pad] in f16 (stride 160)
// ---------------------------------------------------------------------------
__global__ void build_xcat_kernel(const float* __restrict__ x,
                                  const float* __restrict__ S,
                                  const float* __restrict__ T,
                                  const float* __restrict__ mask,
                                  _Float16* __restrict__ xc, int din) {
    int m = blockIdx.x * blockDim.x + threadIdx.x;
    if (m >= M_TOT) return;
    float mk = mask[m];
    float s = S[m] * mk;
    float t = T[m] * mk;
    const float* xr = x + (size_t)m * 64;
    _Float16* o = xc + (size_t)m * XCAT_STRIDE;
    for (int f = 0; f < din; ++f) {
        float xv = xr[f];
        o[f]       = (_Float16)xv;
        o[din + f] = (_Float16)(xv * s - xv);
    }
    o[2 * din] = (_Float16)t;
    for (int f = 2 * din + 1; f < XCAT_STRIDE; ++f) o[f] = (_Float16)0.0f;
}

// ---------------------------------------------------------------------------
// Weight staging: f32 (K x N, row-major) -> f16 TRANSPOSED (N x Kpad),
// zero-padding K up to Kpad. Transposed layout makes the WMMA B-fragment a
// contiguous per-lane run (merges into global_load_b128 pairs).
// ---------------------------------------------------------------------------
__global__ void convert_transpose_kernel(const float* __restrict__ src,
                                         _Float16* __restrict__ dst,
                                         int K, int N, int Kpad) {
    int idx = blockIdx.x * blockDim.x + threadIdx.x;
    int total = N * Kpad;
    if (idx >= total) return;
    int n = idx / Kpad;
    int k = idx % Kpad;
    dst[idx] = (k < K) ? (_Float16)src[(size_t)k * N + n] : (_Float16)0.0f;
}

// Flat f32 -> f16 convert
__global__ void convert_kernel(const float* __restrict__ src,
                               _Float16* __restrict__ dst, int total) {
    int idx = blockIdx.x * blockDim.x + threadIdx.x;
    if (idx >= total) return;
    dst[idx] = (_Float16)src[idx];
}

// ---------------------------------------------------------------------------
// WMMA fragment index helpers (CDNA5 ISA 16-bit layouts, wave32)
//   A 16x32: e=2v+p -> K = (v>=4?16:0) + (half?8:0) + 2*(v&3) + p
//   kernels load A and B alike as K = half*16 + e, not ISA order (with Wt N-major a contiguous run)
// ---------------------------------------------------------------------------
__device__ __forceinline__ int a_k_of(int e, int half_) {
    int v = e >> 1, p = e & 1;
    return ((v & 4) ? 16 : 0) + (half_ ? 8 : 0) + ((v & 3) << 1) + p;
}

// ---------------------------------------------------------------------------
// K3: GNN GEMM  (M=32768, K=Ksteps*32, N=64) + bias + BN + tanh + mask -> f32
// One wave per 16x16 tile; no divergence (EXEC all ones for WMMA).
// ---------------------------------------------------------------------------
__global__ __launch_bounds__(128) void gnn_gemm_kernel(
        const _Float16* __restrict__ A,      // xcat16, stride XCAT_STRIDE
        const _Float16* __restrict__ Wt,     // 64 x ldwt f16 (transposed weights)
        int ldwt,
        const float* __restrict__ bias,      // 64
        const float* __restrict__ gamma,
        const float* __restrict__ beta,
        const float* __restrict__ mask,      // B*N
        float* __restrict__ Xout,            // B*N x 64 f32
        int Ksteps) {
    int wave = (blockIdx.x * blockDim.x + threadIdx.x) >> 5;
    int lane = threadIdx.x & 31;
    const int NT = NOUT / 16;                // 4 tiles in N
    int tm = wave / NT, tn = wave % NT;
    int row0 = tm * 16, col0 = tn * 16;
    int half_ = lane >> 4;
    int lm = lane & 15;
    const _Float16* arow = A  + (size_t)(row0 + lm) * XCAT_STRIDE;
    const _Float16* brow = Wt + (size_t)(col0 + lm) * ldwt + half_ * 16;
    v8f acc = {};
    for (int ks = 0; ks < Ksteps; ++ks) {
        int kb = ks * 32;
        if (ks + 1 < Ksteps) {
            __builtin_prefetch(arow + kb + 32, 0, 3);
            __builtin_prefetch(brow + kb + 32, 0, 3);
        }
        v16h a, bm;
        #pragma unroll
        for (int e = 0; e < 16; ++e) {
            a[e]  = arow[kb + half_ * 16 + e];
            bm[e] = brow[kb + e];
        }
        acc = __builtin_amdgcn_wmma_f32_16x16x32_f16(false, a, false, bm,
                                                     (short)0, acc, false, false);
    }
    const float invs = rsqrtf(1.001f);       // 1/sqrt(1 + BN_EPS)
    #pragma unroll
    for (int r = 0; r < 8; ++r) {
        int m = row0 + r + 8 * half_;
        int f = col0 + lm;
        float v = acc[r] + bias[f];
        v = gamma[f] * v * invs + beta[f];
        v = tanhf(v);
        v *= mask[m];
        Xout[(size_t)m * 64 + f] = v;
    }
}

// ---------------------------------------------------------------------------
// K5/K6: dense GEMM + bias + BN + sigmoid -> f16 out (ld = Nout)
// ---------------------------------------------------------------------------
__global__ __launch_bounds__(128) void dense_gemm_kernel(
        const _Float16* __restrict__ A, int lda,
        const _Float16* __restrict__ Wt, int ldwt,   // Nout x ldwt (transposed)
        const float* __restrict__ bias,
        const float* __restrict__ gamma,
        const float* __restrict__ beta,
        _Float16* __restrict__ Out, int Nout, int Ksteps) {
    int wave = (blockIdx.x * blockDim.x + threadIdx.x) >> 5;
    int lane = threadIdx.x & 31;
    int NT = Nout / 16;
    int tm = wave / NT, tn = wave % NT;
    int row0 = tm * 16, col0 = tn * 16;
    int half_ = lane >> 4;
    int lm = lane & 15;
    const _Float16* arow = A  + (size_t)(row0 + lm) * lda;
    const _Float16* brow = Wt + (size_t)(col0 + lm) * ldwt + half_ * 16;
    v8f acc = {};
    for (int ks = 0; ks < Ksteps; ++ks) {
        int kb = ks * 32;
        if (ks + 1 < Ksteps) {
            __builtin_prefetch(arow + kb + 32, 0, 3);
            __builtin_prefetch(brow + kb + 32, 0, 3);
        }
        v16h a, bm;
        #pragma unroll
        for (int e = 0; e < 16; ++e) {
            a[e]  = arow[kb + half_ * 16 + e];
            bm[e] = brow[kb + e];
        }
        acc = __builtin_amdgcn_wmma_f32_16x16x32_f16(false, a, false, bm,
                                                     (short)0, acc, false, false);
    }
    const float invs = rsqrtf(1.001f);
    #pragma unroll
    for (int r = 0; r < 8; ++r) {
        int m = row0 + r + 8 * half_;
        int f = col0 + lm;
        float v = acc[r] + bias[f];
        v = gamma[f] * v * invs + beta[f];
        v = 1.0f / (1.0f + __expf(-v));
        Out[(size_t)m * Nout + f] = (_Float16)v;
    }
}

// ---------------------------------------------------------------------------
// K7: tiny head: out[b, 0:2] = h1[b] @ W2 + b2 ; out[b, 2:4] = 0
// ---------------------------------------------------------------------------
__global__ void head_kernel(const _Float16* __restrict__ h1,
                            const float* __restrict__ W2,
                            const float* __restrict__ b2,
                            float* __restrict__ out) {
    int tid = blockIdx.x * blockDim.x + threadIdx.x;
    if (tid >= BATCH * 4) return;
    int b = tid >> 2, j = tid & 3;
    float v = 0.0f;
    if (j < 2) {
        v = b2[j];
        for (int k = 0; k < NDENSE; ++k)
            v += (float)h1[b * NDENSE + k] * W2[k * 2 + j];
    }
    out[b * 4 + j] = v;
}

// ---------------------------------------------------------------------------
extern "C" void kernel_launch(void* const* d_in, const int* in_sizes, int n_in,
                              void* d_out, int out_size, void* d_ws, size_t ws_size,
                              hipStream_t stream) {
    const float* xx        = (const float*)d_in[0];
    const float* emb1      = (const float*)d_in[1];
    const float* emb2      = (const float*)d_in[2];
    const float* emb3      = (const float*)d_in[3];
    const float* A0        = (const float*)d_in[4];
    const float* b0        = (const float*)d_in[5];
    const float* A_rest    = (const float*)d_in[6];
    const float* b_rest    = (const float*)d_in[7];
    const float* bn_gamma  = (const float*)d_in[8];
    const float* bn_beta   = (const float*)d_in[9];
    const float* dW0       = (const float*)d_in[10];
    const float* db0       = (const float*)d_in[11];
    const float* dW1       = (const float*)d_in[12];
    const float* db1       = (const float*)d_in[13];
    const float* dbn_gamma = (const float*)d_in[14];
    const float* dbn_beta  = (const float*)d_in[15];
    const float* W2        = (const float*)d_in[16];
    const float* b2        = (const float*)d_in[17];

    char* ws = (char*)d_ws;
    float*     mask   = (float*)(ws + OFF_MASK);
    float*     S      = (float*)(ws + OFF_S);
    float*     T      = (float*)(ws + OFF_T);
    float*     x      = (float*)(ws + OFF_X);
    _Float16*  xcat   = (_Float16*)(ws + OFF_XCAT);
    _Float16*  W16t   = (_Float16*)(ws + OFF_W16);
    _Float16*  xfin16 = (_Float16*)(ws + OFF_XFIN);
    _Float16*  dW0t   = (_Float16*)(ws + OFF_DW0);
    _Float16*  h0_16  = (_Float16*)(ws + OFF_H0);
    _Float16*  dW1t   = (_Float16*)(ws + OFF_DW1);
    _Float16*  h1_16  = (_Float16*)(ws + OFF_H1);

    // K0: embeddings + concat
    prep_kernel<<<(M_TOT + 255) / 256, 256, 0, stream>>>(xx, emb1, emb2, emb3, x, mask);

    // Layer 0: din=33, coords at features 31..32 (last NDIM0=2), K=67 -> pad 96
    pairwise_kernel<<<BATCH, 256, 0, stream>>>(x, 31, 2, S, T);
    build_xcat_kernel<<<(M_TOT + 255) / 256, 256, 0, stream>>>(x, S, T, mask, xcat, 33);
    convert_transpose_kernel<<<(64 * 96 + 255) / 256, 256, 0, stream>>>(A0, W16t, 67, 64, 96);
    gnn_gemm_kernel<<<(M_TOT / 16) * (NOUT / 16) / 4, 128, 0, stream>>>(
        xcat, W16t, 96, b0, bn_gamma + 0, bn_beta + 0, mask, x, 96 / 32);

    // Layers 1..4: din=64, coords at features 60..63 (last NDIM=4), K=129 -> pad 160
    for (int k = 0; k < 4; ++k) {
        pairwise_kernel<<<BATCH, 256, 0, stream>>>(x, 60, 4, S, T);
        build_xcat_kernel<<<(M_TOT + 255) / 256, 256, 0, stream>>>(x, S, T, mask, xcat, 64);
        convert_transpose_kernel<<<(64 * 160 + 255) / 256, 256, 0, stream>>>(
            A_rest + (size_t)k * 129 * 64, W16t, 129, 64, 160);
        gnn_gemm_kernel<<<(M_TOT / 16) * (NOUT / 16) / 4, 128, 0, stream>>>(
            xcat, W16t, 160, b_rest + k * 64, bn_gamma + (k + 1) * 64, bn_beta + (k + 1) * 64,
            mask, x, 160 / 32);
    }

    // Dense head: x(f32, B x 32768 contiguous) -> f16; dW0 -> f16 transposed
    convert_kernel<<<(M_TOT * 64 + 255) / 256, 256, 0, stream>>>(x, xfin16, M_TOT * 64);
    convert_transpose_kernel<<<(NDENSE * (M_TOT * 64 / BATCH) + 255) / 256, 256, 0, stream>>>(
        dW0, dW0t, M_TOT * 64 / BATCH, NDENSE, M_TOT * 64 / BATCH);
    // GEMM0: M=64, K=32768, N=128 -> sigmoid(bn(...)) f16
    dense_gemm_kernel<<<(64 / 16) * (NDENSE / 16) / 4, 128, 0, stream>>>(
        xfin16, M_TOT * 64 / BATCH, dW0t, M_TOT * 64 / BATCH,
        db0, dbn_gamma + 0, dbn_beta + 0, h0_16, NDENSE, (M_TOT * 64 / BATCH) / 32);
    // GEMM1: M=64, K=128, N=128
    convert_transpose_kernel<<<(NDENSE * NDENSE + 255) / 256, 256, 0, stream>>>(
        dW1, dW1t, NDENSE, NDENSE, NDENSE);
    dense_gemm_kernel<<<(64 / 16) * (NDENSE / 16) / 4, 128, 0, stream>>>(
        h0_16, NDENSE, dW1t, NDENSE,
        db1, dbn_gamma + NDENSE, dbn_beta + NDENSE, h1_16, NDENSE, NDENSE / 32);

    // Final 128x2 head + zero padding columns
    head_kernel<<<1, 256, 0, stream>>>(h1_16, W2, b2, (float*)d_out);
}
